// PointNextLangHierachicalEncoder_64510408786140
// MI455X (gfx1250) — matrix operand order repaired and checked by reading, not compiled, not run on hardware
//
#include <hip/hip_runtime.h>

typedef _Float16 v16h __attribute__((ext_vector_type(16)));
typedef float    v8f  __attribute__((ext_vector_type(8)));

#define NSAMP 32

__device__ __forceinline__ v8f wmma16x16x32(v16h a, v16h b, v8f c) {
  // D = A(16x32 f16) * B(32x16 f16) + C(16x16 f32)
  return __builtin_amdgcn_wmma_f32_16x16x32_f16(false, a, false, b, (short)0, c,
                                                false, false);
}

// ---------------------------------------------------------------------------
// Pack weight matrix W(O x Kact, f32, row-major) into an A-fragment order with contiguous k per lane half:
//   Wp[((ot*KC + kc)*32 + lane)*16 + e] = f16( W[ot*16 + lane%16][ch] )
//   ch = kc*32 + (lane>=16)*16 + e ; zero-padded for ch >= Kact.
// One contiguous 32B load per lane then yields a ready A fragment.
// ---------------------------------------------------------------------------
__global__ void pack_weights_kernel(const float* __restrict__ W,
                                    _Float16* __restrict__ Wp,
                                    int Kact, int Kp, int total) {
  int t = blockIdx.x * blockDim.x + threadIdx.x;
  if (t >= total) return;  // total = O * Kp
  int e = t & 15;
  int r = t >> 4;
  int lane = r & 31;
  int r2 = r >> 5;
  int KC = Kp >> 5;
  int kc = r2 % KC;
  int ot = r2 / KC;
  int orow = ot * 16 + (lane & 15);
  int ch = kc * 32 + ((lane >> 4) & 1) * 16 + e;
  float v = (ch < Kact) ? W[(size_t)orow * Kact + ch] : 0.0f;
  Wp[t] = (_Float16)v;
}

// ---------------------------------------------------------------------------
// Pad stem input (B,4,N) -> (B,32,N) with zero channels, so pw_conv needs no
// bounds checks anywhere.
// ---------------------------------------------------------------------------
__global__ void pad_stem_kernel(const float* __restrict__ fin,
                                float* __restrict__ out, int N) {
  int t = blockIdx.x * blockDim.x + threadIdx.x;
  if (t >= 8 * 32 * N) return;
  int n = t % N;
  int r = t / N;
  int ch = r & 31;
  int b = r >> 5;
  out[t] = (ch < 4) ? fin[((size_t)b * 4 + ch) * N + n] : 0.0f;
}

// ---------------------------------------------------------------------------
// Furthest point sampling: one block per batch, serial chain with LDS argmax.
// ---------------------------------------------------------------------------
__global__ void __launch_bounds__(256) fps_kernel(const float* __restrict__ p,
                                                  float* __restrict__ dist,
                                                  int* __restrict__ idx,
                                                  int N, int M) {
  const int T = 256;
  int b = blockIdx.x;
  int tid = threadIdx.x;
  __shared__ float sval[256];
  __shared__ int sidx[256];
  __shared__ float lastp[3];
  __shared__ int slast;

  const float* pb = p + (size_t)b * N * 3;
  float* db = dist + (size_t)b * N;
  for (int n = tid; n < N; n += T) db[n] = 1e10f;
  if (tid == 0) { idx[(size_t)b * M] = 0; slast = 0; }
  __syncthreads();

  for (int i = 1; i < M; ++i) {
    if (tid < 3) lastp[tid] = pb[(size_t)slast * 3 + tid];
    __syncthreads();
    float lx = lastp[0], ly = lastp[1], lz = lastp[2];
    float best = -1.0f;
    int bi = 0;
    for (int n = tid; n < N; n += T) {
      float dx = pb[n * 3 + 0] - lx;
      float dy = pb[n * 3 + 1] - ly;
      float dz = pb[n * 3 + 2] - lz;
      float d = dx * dx + dy * dy + dz * dz;
      float nd = fminf(db[n], d);
      db[n] = nd;
      if (nd > best) { best = nd; bi = n; }
    }
    sval[tid] = best;
    sidx[tid] = bi;
    __syncthreads();
    for (int s = 128; s > 0; s >>= 1) {
      if (tid < s) {
        float ov = sval[tid + s]; int oi = sidx[tid + s];
        if (ov > sval[tid] || (ov == sval[tid] && oi < sidx[tid])) {
          sval[tid] = ov; sidx[tid] = oi;
        }
      }
      __syncthreads();
    }
    if (tid == 0) { idx[(size_t)b * M + i] = sidx[0]; slast = sidx[0]; }
    __syncthreads();
  }
}

// ---------------------------------------------------------------------------
__global__ void gather_points_kernel(const float* __restrict__ p,
                                     const int* __restrict__ idx,
                                     float* __restrict__ out, int N, int M) {
  int t = blockIdx.x * blockDim.x + threadIdx.x;
  if (t >= 8 * M) return;
  int b = t / M, m = t - b * M;
  int n = idx[(size_t)b * M + m];
  out[((size_t)b * M + m) * 3 + 0] = p[((size_t)b * N + n) * 3 + 0];
  out[((size_t)b * M + m) * 3 + 1] = p[((size_t)b * N + n) * 3 + 1];
  out[((size_t)b * M + m) * 3 + 2] = p[((size_t)b * N + n) * 3 + 2];
}

// ---------------------------------------------------------------------------
// Ball query: first NSAMP support indices (in order) within radius, padded
// with the first valid neighbor (CUDA ball_query semantics).
// ---------------------------------------------------------------------------
__global__ void ball_query_kernel(const float* __restrict__ sup,
                                  const float* __restrict__ qry,
                                  int* __restrict__ nbr,
                                  int N, int M, float r2) {
  int t = blockIdx.x * blockDim.x + threadIdx.x;
  if (t >= 8 * M) return;
  int b = t / M, m = t - b * M;
  float qx = qry[((size_t)b * M + m) * 3 + 0];
  float qy = qry[((size_t)b * M + m) * 3 + 1];
  float qz = qry[((size_t)b * M + m) * 3 + 2];
  int* o = nbr + ((size_t)b * M + m) * NSAMP;
  int cnt = 0;
  for (int n = 0; n < N && cnt < NSAMP; ++n) {
    float dx = sup[((size_t)b * N + n) * 3 + 0] - qx;
    float dy = sup[((size_t)b * N + n) * 3 + 1] - qy;
    float dz = sup[((size_t)b * N + n) * 3 + 2] - qz;
    if (dx * dx + dy * dy + dz * dz < r2) o[cnt++] = n;
  }
  if (cnt == 0) { o[0] = 0; cnt = 1; }
  int f0 = o[0];
  for (int j = cnt; j < NSAMP; ++j) o[j] = f0;
}

// ---------------------------------------------------------------------------
// Grouped conv + relu + max over K neighbors, fused, WMMA f16->f32.
// fj gathered to LDS column-major [col][Kp]; A from pre-swizzled weights.
// Inner loop: 2x global b128 (A) + 4x ds b128 (B) + 2x v_wmma. No branches.
// ---------------------------------------------------------------------------
__global__ void __launch_bounds__(256) sa_conv_max_kernel(
    const float* __restrict__ psup, const float* __restrict__ pq,
    const float* __restrict__ fin, const int* __restrict__ nbr,
    const _Float16* __restrict__ Wp, const float* __restrict__ bias,
    float* __restrict__ out,
    int N, int M, int Cin, int Cout, int Kp, int MTILE) {
  __shared__ __align__(64) _Float16 FJ[32768];  // 64 KB
  int mtiles = M / MTILE;
  int b = blockIdx.x / mtiles;
  int m0 = (blockIdx.x - b * mtiles) * MTILE;
  int tid = threadIdx.x;
  int total = MTILE * NSAMP * Kp;

  for (int t = tid; t < total; t += 256) {
    int col = t / Kp;
    int row = t - col * Kp;
    int ml = col >> 5;
    int k = col & 31;
    int m = m0 + ml;
    int n = nbr[((size_t)b * M + m) * NSAMP + k];
    float v = 0.0f;
    if (row < 3)
      v = psup[((size_t)b * N + n) * 3 + row] - pq[((size_t)b * M + m) * 3 + row];
    else if (row < 3 + Cin)
      v = fin[((size_t)b * Cin + (row - 3)) * N + n];
    FJ[col * Kp + row] = (_Float16)v;
  }
  __syncthreads();

  int wave = tid >> 5;
  int lane = tid & 31;
  int lane16 = lane & 15;
  int hi = (lane >> 4) & 1;
  int otiles = Cout >> 4;
  int tasks = otiles * MTILE;
  int KC = Kp >> 5;
  const v16h* WA = (const v16h*)Wp;

  for (int task = wave; task < tasks; task += 8) {
    int ot = task / MTILE;
    int ml = task - ot * MTILE;
    v8f acc0 = {0, 0, 0, 0, 0, 0, 0, 0};
    v8f acc1 = {0, 0, 0, 0, 0, 0, 0, 0};
    for (int kc = 0; kc < KC; ++kc) {
      v16h a = WA[((size_t)ot * KC + kc) * 32 + lane];
      int koff = (kc << 5) + hi * 16;
      v16h b0 = *(const v16h*)(FJ + (ml * NSAMP + lane16) * Kp + koff);
      v16h b1 = *(const v16h*)(FJ + (ml * NSAMP + 16 + lane16) * Kp + koff);
      acc0 = wmma16x16x32(a, b0, acc0);
      acc1 = wmma16x16x32(a, b1, acc1);
    }
    // epilogue: bias + relu + max over the 32 neighbor columns
    int m = m0 + ml;
#pragma unroll
    for (int r = 0; r < 8; ++r) {
      int row = r + hi * 8;
      float bv = bias[ot * 16 + row];
      float v = fmaxf(fmaxf(acc0[r] + bv, 0.0f), fmaxf(acc1[r] + bv, 0.0f));
      v = fmaxf(v, __shfl_xor(v, 1, 32));
      v = fmaxf(v, __shfl_xor(v, 2, 32));
      v = fmaxf(v, __shfl_xor(v, 4, 32));
      v = fmaxf(v, __shfl_xor(v, 8, 32));
      if (lane16 == 0)
        out[((size_t)b * Cout + ot * 16 + row) * M + m] = v;
    }
  }
}

// ---------------------------------------------------------------------------
// Pointwise conv GEMM via WMMA: out(B,O,M) = W(O,C) @ X(B,C,M) + bias.
// C is always a multiple of 32 (stem input is pre-padded), so the B-fragment
// loop is 16 unguarded coalesced loads. A from pre-swizzled weights.
// mode 0: bias; 1: bias+relu; 2: bias+residual+relu.
// ---------------------------------------------------------------------------
__global__ void __launch_bounds__(256) pw_conv_kernel(
    const _Float16* __restrict__ Wp, const float* __restrict__ bias,
    const float* __restrict__ X, const float* __restrict__ res,
    float* __restrict__ out,
    int C, int O, int M, int mode, int tiles) {
  int wave = threadIdx.x >> 5;
  int lane = threadIdx.x & 31;
  int lane16 = lane & 15;
  int hi = (lane >> 4) & 1;
  int gw = blockIdx.x * 8 + wave;
  if (gw >= tiles) return;
  int mt_n = M >> 4;
  int pb = (O >> 4) * mt_n;
  int b = gw / pb;
  int rtile = gw - b * pb;
  int ot = rtile / mt_n;
  int mt = rtile - ot * mt_n;
  int m = mt * 16 + lane16;
  int KC = C >> 5;
  const v16h* WA = (const v16h*)Wp;

  v8f acc = {0, 0, 0, 0, 0, 0, 0, 0};
  for (int kc = 0; kc < KC; ++kc) {
    v16h a = WA[((size_t)ot * KC + kc) * 32 + lane];
    v16h bf;
    int krow0 = (kc << 5) + hi * 16;
    const float* xp = X + ((size_t)b * C + krow0) * M + m;
#pragma unroll
    for (int e = 0; e < 16; ++e)
      bf[e] = (_Float16)xp[(size_t)e * M];
    acc = wmma16x16x32(a, bf, acc);
  }
#pragma unroll
  for (int r = 0; r < 8; ++r) {
    int row = r + hi * 8;
    int o = ot * 16 + row;
    float v = acc[r] + bias[o];
    if (mode == 2) v += res[((size_t)b * O + o) * M + m];
    if (mode >= 1) v = fmaxf(v, 0.0f);
    out[((size_t)b * O + o) * M + m] = v;
  }
}

// ---------------------------------------------------------------------------
extern "C" void kernel_launch(void* const* d_in, const int* in_sizes, int n_in,
                              void* d_out, int out_size, void* d_ws, size_t ws_size,
                              hipStream_t stream) {
  (void)n_in; (void)out_size; (void)ws_size;

  // Input mapping: layout A = dict insertion order, layout B = sorted pytree.
  const float* p_in;
  const float* f_in;
  const float* stemW;
  const float* stemB;
  const float* SP[5][8];  // per stage: saW,saB,laW,laB,pw1W,pw1B,pw2W,pw2B
  if (in_sizes[0] == 8 * 8192 * 3) {
    p_in = (const float*)d_in[0];
    f_in = (const float*)d_in[1];
    stemW = (const float*)d_in[2];
    stemB = (const float*)d_in[3];
    for (int s = 1; s <= 4; ++s)
      for (int j = 0; j < 8; ++j)
        SP[s][j] = (const float*)d_in[4 + (s - 1) * 8 + j];
  } else {
    f_in = (const float*)d_in[0];
    p_in = (const float*)d_in[1];
    static const int remap[8] = {6, 7, 0, 1, 2, 3, 4, 5};
    for (int s = 1; s <= 4; ++s)
      for (int j = 0; j < 8; ++j)
        SP[s][j] = (const float*)d_in[2 + (s - 1) * 8 + remap[j]];
    stemW = (const float*)d_in[34];
    stemB = (const float*)d_in[35];
  }

  const int CH[5] = {32, 64, 128, 256, 512};
  const float RSA[5] = {0.f, 0.1f, 0.2f, 0.4f, 0.8f};
  const float RIR[5] = {0.f, 0.2f, 0.4f, 0.8f, 1.6f};
  const int MTSA[5] = {0, 16, 8, 4, 2};       // queries per block (SA)
  const int KPSA[5] = {0, 64, 96, 160, 288};  // padded K-dim (SA)
  const int MTLA[5] = {0, 8, 4, 2, 1};        // queries per block (LA)
  const int KPLA[5] = {0, 96, 160, 288, 544};

  // Workspace layout (floats)
  float* ws = (float*)d_ws;
  size_t off = 0;
  float* pA = ws + off; off += (size_t)8 * 2048 * 3;
  float* pB = ws + off; off += (size_t)8 * 2048 * 3;
  float* f0 = ws + off; off += (size_t)8 * 32 * 8192;
  float* f1 = ws + off; off += (size_t)8 * 32 * 8192;
  float* f2 = ws + off; off += (size_t)8 * 32 * 8192;
  float* hb = ws + off; off += (size_t)8 * 256 * 2048;
  float* dist = ws + off; off += (size_t)8 * 8192;
  int* fpsidx = (int*)(ws + off); off += (size_t)8 * 2048;
  int* nbr = (int*)(ws + off); off += (size_t)8 * 2048 * 32;
  _Float16* wpk = (_Float16*)(ws + off);  // packed weights (halves)

  // --- Pack all weights into A-fragment order (17 matrices) ---
  _Float16* stemWp;
  _Float16* SWp[5][4];  // sa, la, pw1, pw2
  {
    size_t hoff = 0;
    auto pack = [&](const float* W, int O, int Kact, int Kp) -> _Float16* {
      _Float16* dst = wpk + hoff;
      int total = O * Kp;
      hoff += (size_t)total;
      pack_weights_kernel<<<(total + 255) / 256, 256, 0, stream>>>(W, dst, Kact, Kp, total);
      return dst;
    };
    stemWp = pack(stemW, 32, 4, 32);
    for (int s = 1; s <= 4; ++s) {
      int c = CH[s], cin = CH[s - 1];
      SWp[s][0] = pack(SP[s][0], c, cin + 3, KPSA[s]);
      SWp[s][1] = pack(SP[s][2], c, c + 3, KPLA[s]);
      SWp[s][2] = pack(SP[s][4], 4 * c, c, c);
      SWp[s][3] = pack(SP[s][6], c, 4 * c, 4 * c);
    }
  }

  const int Bn = 8;

  // --- Stem: pad input to 32 channels (into f2), then 1x1 conv -> f0 ---
  {
    int total = 8 * 32 * 8192;
    pad_stem_kernel<<<(total + 255) / 256, 256, 0, stream>>>(f_in, f2, 8192);
    int O = 32, M = 8192;
    int tiles = Bn * (O / 16) * (M / 16);
    pw_conv_kernel<<<(tiles + 7) / 8, 256, 0, stream>>>(
        stemWp, stemB, f2, nullptr, f0, 32, O, M, 0, tiles);
  }

  const float* pcur = p_in;
  float* fcur = f0;
  int N = 8192;

  for (int s = 1; s <= 4; ++s) {
    int M = N / 4;
    int c = CH[s], cin = CH[s - 1];
    float* pnew = (s & 1) ? pA : pB;
    int thr = Bn * M;

    // --- SetAbstraction ---
    fps_kernel<<<Bn, 256, 0, stream>>>(pcur, dist, fpsidx, N, M);
    gather_points_kernel<<<(thr + 255) / 256, 256, 0, stream>>>(pcur, fpsidx, pnew, N, M);
    ball_query_kernel<<<(thr + 255) / 256, 256, 0, stream>>>(
        pcur, pnew, nbr, N, M, RSA[s] * RSA[s]);
    sa_conv_max_kernel<<<Bn * (M / MTSA[s]), 256, 0, stream>>>(
        pcur, pnew, fcur, nbr, SWp[s][0], SP[s][1], f1,
        N, M, cin, c, KPSA[s], MTSA[s]);

    // --- InvResMLP: LocalAggregation ---
    ball_query_kernel<<<(thr + 255) / 256, 256, 0, stream>>>(
        pnew, pnew, nbr, M, M, RIR[s] * RIR[s]);
    sa_conv_max_kernel<<<Bn * (M / MTLA[s]), 256, 0, stream>>>(
        pnew, pnew, f1, nbr, SWp[s][1], SP[s][3], f2,
        M, M, c, c, KPLA[s], MTLA[s]);

    // --- pw1 (expansion 4, relu) ---
    {
      int tiles = Bn * ((4 * c) / 16) * (M / 16);
      pw_conv_kernel<<<(tiles + 7) / 8, 256, 0, stream>>>(
          SWp[s][2], SP[s][5], f2, nullptr, hb, c, 4 * c, M, 1, tiles);
    }
    // --- pw2 + residual + relu ---
    {
      float* oout = (s == 4) ? (float*)d_out : f0;
      int tiles = Bn * (c / 16) * (M / 16);
      pw_conv_kernel<<<(tiles + 7) / 8, 256, 0, stream>>>(
          SWp[s][3], SP[s][7], hb, f1, oout, 4 * c, c, M, 2, tiles);
    }

    pcur = pnew;
    fcur = f0;
    N = M;
  }
}
